// RWKV_Tmix_x070_72181220376607
// MI455X (gfx1250) — compile-verified, matrix operand order checked by reading, not run on hardware
//
#include <hip/hip_runtime.h>
#include <hip/hip_bf16.h>
#include <math.h>

// ---------------------------------------------------------------------------
// Types for CDNA5 WMMA (wave32)
// ---------------------------------------------------------------------------
typedef __attribute__((ext_vector_type(16))) __bf16 v16bf;
typedef __attribute__((ext_vector_type(8)))  __bf16 v8bf;
typedef __attribute__((ext_vector_type(8)))  float  v8f;

#define B_ 2
#define T_ 1024
#define C_ 2048
#define H_ 32
#define HEAD_ 64
#define BT_ (B_ * T_)
#define BTC_ (BT_ * C_)
#define GN_EPS 0.00064f

// ---------------------------------------------------------------------------
// Transpose + fp32 -> bf16 convert:  in [K,N] f32  ->  out [N,K] bf16
// ---------------------------------------------------------------------------
__global__ void k_transpose_bf16(const float* __restrict__ in,
                                 __bf16* __restrict__ out, int K, int N) {
  int tot = K * N;
  for (int i = blockIdx.x * blockDim.x + threadIdx.x; i < tot;
       i += gridDim.x * blockDim.x) {
    int n = i / K;
    int k = i - n * K;
    out[(size_t)n * K + k] = (__bf16)in[(size_t)k * N + n];
  }
}

// ---------------------------------------------------------------------------
// Token-shift mixing: xx = shift(x)-x ; out_i = bf16(x + xx*coef_i)
// ---------------------------------------------------------------------------
__global__ void k_mix(const float* __restrict__ x,
                      const float* __restrict__ cr, const float* __restrict__ cw,
                      const float* __restrict__ ck, const float* __restrict__ cv,
                      const float* __restrict__ ca, const float* __restrict__ cg,
                      __bf16* __restrict__ xr, __bf16* __restrict__ xw,
                      __bf16* __restrict__ xk, __bf16* __restrict__ xv,
                      __bf16* __restrict__ xa, __bf16* __restrict__ xg) {
  for (int i = blockIdx.x * blockDim.x + threadIdx.x; i < BTC_;
       i += gridDim.x * blockDim.x) {
    int c = i & (C_ - 1);
    int bt = i / C_;
    int t = bt & (T_ - 1);
    float xi = x[i];
    float prev = (t > 0) ? x[i - C_] : 0.f;
    float xx = prev - xi;
    xr[i] = (__bf16)(xi + xx * cr[c]);
    xw[i] = (__bf16)(xi + xx * cw[c]);
    xk[i] = (__bf16)(xi + xx * ck[c]);
    xv[i] = (__bf16)(xi + xx * cv[c]);
    xa[i] = (__bf16)(xi + xx * ca[c]);
    xg[i] = (__bf16)(xi + xx * cg[c]);
  }
}

// ---------------------------------------------------------------------------
// Activation + fp32 -> bf16.  MODE: 0=identity, 1=tanh, 2=sigmoid
// ---------------------------------------------------------------------------
template <int MODE>
__global__ void k_act_bf16(const float* __restrict__ in,
                           __bf16* __restrict__ out, int n) {
  for (int i = blockIdx.x * blockDim.x + threadIdx.x; i < n;
       i += gridDim.x * blockDim.x) {
    float v = in[i];
    if (MODE == 1) v = tanhf(v);
    if (MODE == 2) v = 1.f / (1.f + expf(-v));
    out[i] = (__bf16)v;
  }
}

// ---------------------------------------------------------------------------
// Fragment load per ISA 7.12.2 (16-bit A 16x32): lanes 0-15 -> K base..+7 and
// +16..+23, lanes 16-31 -> +8..+15 and +24..+31; B mirrors with N as row.
// ---------------------------------------------------------------------------
__device__ __forceinline__ v16bf load_frag(const __bf16* __restrict__ p) {
  v8bf lo = *reinterpret_cast<const v8bf*>(p);
  v8bf hi = *reinterpret_cast<const v8bf*>(p + 16);
  v16bf f;
#pragma unroll
  for (int i = 0; i < 8; ++i) { f[i] = lo[i]; f[i + 8] = hi[i]; }
  return f;
}

// ---------------------------------------------------------------------------
// Generic WMMA bf16 GEMM, one wave per 32x32 tile (used for skinny N).
// C[M,N] = A[M,K] @ B[K,N], with B transposed as Bt[N,K].
// ---------------------------------------------------------------------------
__global__ __launch_bounds__(32) void k_gemm_bf16_32(
    const __bf16* __restrict__ A, const __bf16* __restrict__ Bt,
    float* __restrict__ C, int M, int N, int K) {
  const int lane = threadIdx.x & 31;
  const int half = lane >> 4;
  const int l16 = lane & 15;
  const int koff = half << 3;
  const int m0 = blockIdx.y << 5;
  const int n0 = blockIdx.x << 5;

  v8f acc[2][2];
#pragma unroll
  for (int mi = 0; mi < 2; ++mi)
#pragma unroll
    for (int ni = 0; ni < 2; ++ni)
#pragma unroll
      for (int e = 0; e < 8; ++e) acc[mi][ni][e] = 0.f;

  const __bf16* a0p = A + (size_t)(m0 + l16) * K + koff;
  const __bf16* a1p = A + (size_t)(m0 + 16 + l16) * K + koff;
  const __bf16* b0p = Bt + (size_t)(n0 + l16) * K + koff;
  const __bf16* b1p = Bt + (size_t)(n0 + 16 + l16) * K + koff;

  for (int k0 = 0; k0 < K; k0 += 32) {
    v16bf a0 = load_frag(a0p + k0);
    v16bf a1 = load_frag(a1p + k0);
    v16bf b0 = load_frag(b0p + k0);
    v16bf b1 = load_frag(b1p + k0);
    acc[0][0] = __builtin_amdgcn_wmma_f32_16x16x32_bf16(
        false, a0, false, b0, (short)0, acc[0][0], false, false);
    acc[0][1] = __builtin_amdgcn_wmma_f32_16x16x32_bf16(
        false, a0, false, b1, (short)0, acc[0][1], false, false);
    acc[1][0] = __builtin_amdgcn_wmma_f32_16x16x32_bf16(
        false, a1, false, b0, (short)0, acc[1][0], false, false);
    acc[1][1] = __builtin_amdgcn_wmma_f32_16x16x32_bf16(
        false, a1, false, b1, (short)0, acc[1][1], false, false);
  }

#pragma unroll
  for (int mi = 0; mi < 2; ++mi)
#pragma unroll
    for (int ni = 0; ni < 2; ++ni)
#pragma unroll
      for (int e = 0; e < 8; ++e) {
        int row = m0 + (mi << 4) + e + (half << 3);
        int col = n0 + (ni << 4) + l16;
        C[(size_t)row * N + col] = acc[mi][ni][e];
      }
}

// ---------------------------------------------------------------------------
// Wide WMMA bf16 GEMM, one wave per 32x64 tile (2x4 x 16x16), 8 WMMA/K-step.
// ~21 FLOP per byte moved from cache; prefetch 4 K-steps ahead (global_prefetch).
// ---------------------------------------------------------------------------
__global__ __launch_bounds__(32) void k_gemm_bf16_2x4(
    const __bf16* __restrict__ A, const __bf16* __restrict__ Bt,
    float* __restrict__ C, int M, int N, int K) {
  const int lane = threadIdx.x & 31;
  const int half = lane >> 4;
  const int l16 = lane & 15;
  const int koff = half << 3;
  const int m0 = blockIdx.y << 5;  // 32 rows
  const int n0 = blockIdx.x << 6;  // 64 cols

  v8f acc[2][4];
#pragma unroll
  for (int mi = 0; mi < 2; ++mi)
#pragma unroll
    for (int ni = 0; ni < 4; ++ni)
#pragma unroll
      for (int e = 0; e < 8; ++e) acc[mi][ni][e] = 0.f;

  const __bf16* ap[2];
  const __bf16* bp[4];
#pragma unroll
  for (int mi = 0; mi < 2; ++mi)
    ap[mi] = A + (size_t)(m0 + mi * 16 + l16) * K + koff;
#pragma unroll
  for (int ni = 0; ni < 4; ++ni)
    bp[ni] = Bt + (size_t)(n0 + ni * 16 + l16) * K + koff;

  for (int k0 = 0; k0 < K; k0 += 32) {
    if (k0 + 128 < K) {  // look ahead beyond the compiler's unroll window
      __builtin_prefetch(ap[0] + k0 + 128, 0, 3);
      __builtin_prefetch(ap[1] + k0 + 128, 0, 3);
      __builtin_prefetch(bp[0] + k0 + 128, 0, 3);
      __builtin_prefetch(bp[2] + k0 + 128, 0, 3);
    }
    v16bf a[2], b[4];
#pragma unroll
    for (int mi = 0; mi < 2; ++mi) a[mi] = load_frag(ap[mi] + k0);
#pragma unroll
    for (int ni = 0; ni < 4; ++ni) b[ni] = load_frag(bp[ni] + k0);
#pragma unroll
    for (int mi = 0; mi < 2; ++mi)
#pragma unroll
      for (int ni = 0; ni < 4; ++ni)
        acc[mi][ni] = __builtin_amdgcn_wmma_f32_16x16x32_bf16(
            false, a[mi], false, b[ni], (short)0, acc[mi][ni], false, false);
  }

#pragma unroll
  for (int mi = 0; mi < 2; ++mi)
#pragma unroll
    for (int ni = 0; ni < 4; ++ni)
#pragma unroll
      for (int e = 0; e < 8; ++e) {
        int row = m0 + (mi << 4) + e + (half << 3);
        int col = n0 + (ni << 4) + l16;
        C[(size_t)row * N + col] = acc[mi][ni][e];
      }
}

// ---------------------------------------------------------------------------
// 64-thread block reduction helper
// ---------------------------------------------------------------------------
__device__ __forceinline__ float blk_reduce64(float v, float* sh, int i) {
  sh[i] = v;
  __syncthreads();
#pragma unroll
  for (int s = 32; s > 0; s >>= 1) {
    if (i < s) sh[i] += sh[i + s];
    __syncthreads();
  }
  float r = sh[0];
  __syncthreads();
  return r;
}

// ---------------------------------------------------------------------------
// Post kernel: one block per (b,t,h), 64 threads (one per channel in head).
// ---------------------------------------------------------------------------
__global__ __launch_bounds__(64) void k_post(
    const float* __restrict__ v0, const float* __restrict__ dw,
    const float* __restrict__ da, const float* __restrict__ dv,
    const float* __restrict__ ww_b, const float* __restrict__ aa_b,
    const float* __restrict__ vv_b, const float* __restrict__ kk_s,
    const float* __restrict__ ka_s, float* __restrict__ kbuf,
    float* __restrict__ vbuf, float* __restrict__ decay,
    float* __restrict__ arec, float* __restrict__ brec) {
  __shared__ float sh[64];
  const int h = blockIdx.x & (H_ - 1);
  const int bt = blockIdx.x >> 5;  // H_ == 32
  const int i = threadIdx.x;
  const int c = h * HEAD_ + i;
  const size_t g = (size_t)bt * C_ + c;

  float kraw = kbuf[g];
  float kk = kraw * kk_s[c];
  float ss = blk_reduce64(kk * kk, sh, i);
  float nrm = fmaxf(sqrtf(ss), 1e-12f);
  float kkn = kk / nrm;

  float a = 1.f / (1.f + expf(-(aa_b[c] + da[g])));
  float vm = 1.f / (1.f + expf(-(vv_b[c] + dv[g])));
  float vraw = vbuf[g];
  float vfin = vraw + (v0[g] - vraw) * vm;

  float z = -(ww_b[c] + dw[g]);
  float sp = (z > 20.f) ? z : log1pf(expf(z));  // softplus(z), stable
  float wraw = -sp - 0.5f;
  float dec = expf(-expf(wraw));

  kbuf[g] = kraw * (1.f + (a - 1.f) * ka_s[c]);
  vbuf[g] = vfin;
  decay[g] = dec;
  arec[g] = -kkn;
  brec[g] = kkn * a;
}

// ---------------------------------------------------------------------------
// RWKV7 recurrence: one 64-thread block per (b,h); thread i owns state row i
// (64 floats in VGPRs); per-step vectors staged through LDS.
// ---------------------------------------------------------------------------
__global__ __launch_bounds__(64) void k_rwkv7(
    const float* __restrict__ r, const float* __restrict__ w,
    const float* __restrict__ k, const float* __restrict__ v,
    const float* __restrict__ a, const float* __restrict__ b,
    float* __restrict__ y) {
  __shared__ float sr[64], sw[64], sk[64], sv[64], sa[64], sb[64];
  const int bh = blockIdx.x;
  const int b_idx = bh / H_;
  const int h = bh - b_idx * H_;
  const int i = threadIdx.x;

  float st[64];
#pragma unroll
  for (int j = 0; j < 64; ++j) st[j] = 0.f;

  size_t base = (size_t)b_idx * T_ * C_ + h * HEAD_;
  for (int t = 0; t < T_; ++t, base += C_) {
    sr[i] = r[base + i];
    sw[i] = w[base + i];
    sk[i] = k[base + i];
    sv[i] = v[base + i];
    sa[i] = a[base + i];
    sb[i] = b[base + i];
    __syncthreads();

    float sdota = 0.f;
#pragma unroll
    for (int j = 0; j < 64; ++j) sdota += st[j] * sa[j];

    float vi = sv[i];
    float out = 0.f;
#pragma unroll
    for (int j = 0; j < 64; ++j) {
      st[j] = st[j] * sw[j] + sdota * sb[j] + vi * sk[j];
      out += st[j] * sr[j];
    }
    y[base + i] = out;
    __syncthreads();
  }
}

// ---------------------------------------------------------------------------
// GroupNorm + per-head bonus + gate -> bf16 input of final GEMM.
// ---------------------------------------------------------------------------
__global__ __launch_bounds__(64) void k_gn_bonus_gate(
    const float* __restrict__ y, const float* __restrict__ r,
    const float* __restrict__ k, const float* __restrict__ v,
    const float* __restrict__ gate, const float* __restrict__ rk_s,
    const float* __restrict__ gn_w, const float* __restrict__ gn_b,
    __bf16* __restrict__ yg) {
  __shared__ float sh[64];
  const int h = blockIdx.x & (H_ - 1);
  const int bt = blockIdx.x >> 5;
  const int i = threadIdx.x;
  const int c = h * HEAD_ + i;
  const size_t g = (size_t)bt * C_ + c;

  float yi = y[g];
  float mu = blk_reduce64(yi, sh, i) * (1.f / HEAD_);
  float d = yi - mu;
  float var = blk_reduce64(d * d, sh, i) * (1.f / HEAD_);
  float gn = d * rsqrtf(var + GN_EPS) * gn_w[c] + gn_b[c];

  float dot = blk_reduce64(r[g] * k[g] * rk_s[h * HEAD_ + i], sh, i);
  float out = gn + dot * v[g];
  yg[g] = (__bf16)(out * gate[g]);
}

__global__ void k_copy_f32(const float* __restrict__ in,
                           float* __restrict__ out, int n) {
  for (int i = blockIdx.x * blockDim.x + threadIdx.x; i < n;
       i += gridDim.x * blockDim.x)
    out[i] = in[i];
}

// ---------------------------------------------------------------------------
// Launch
// ---------------------------------------------------------------------------
extern "C" void kernel_launch(void* const* d_in, const int* in_sizes, int n_in,
                              void* d_out, int out_size, void* d_ws,
                              size_t ws_size, hipStream_t stream) {
  (void)in_sizes; (void)n_in; (void)out_size; (void)ws_size;
  const float* x     = (const float*)d_in[0];
  const float* v0    = (const float*)d_in[1];
  const float* xx_r  = (const float*)d_in[2];
  const float* xx_w  = (const float*)d_in[3];
  const float* xx_k  = (const float*)d_in[4];
  const float* xx_v  = (const float*)d_in[5];
  const float* xx_a  = (const float*)d_in[6];
  const float* xx_g  = (const float*)d_in[7];
  const float* ww_b  = (const float*)d_in[8];
  const float* ww_w1 = (const float*)d_in[9];
  const float* ww_w2 = (const float*)d_in[10];
  const float* aa_b  = (const float*)d_in[11];
  const float* aa_w1 = (const float*)d_in[12];
  const float* aa_w2 = (const float*)d_in[13];
  const float* vv_b  = (const float*)d_in[14];
  const float* vv_w1 = (const float*)d_in[15];
  const float* vv_w2 = (const float*)d_in[16];
  const float* gg_w1 = (const float*)d_in[17];
  const float* gg_w2 = (const float*)d_in[18];
  const float* kk_s  = (const float*)d_in[19];
  const float* ka_s  = (const float*)d_in[20];
  const float* rk_s  = (const float*)d_in[21];
  const float* W_r   = (const float*)d_in[22];
  const float* W_k   = (const float*)d_in[23];
  const float* W_v   = (const float*)d_in[24];
  const float* W_o   = (const float*)d_in[25];
  const float* gn_w  = (const float*)d_in[26];
  const float* gn_b  = (const float*)d_in[27];

  const int D_DECAY = 96, D_AAA = 96, D_MV = 64, D_GATE = 256;

  // bump allocator over d_ws
  char* p = (char*)d_ws;
  auto alloc = [&](size_t bytes) -> void* {
    void* r = (void*)p;
    p += (bytes + 255) & ~(size_t)255;
    return r;
  };

  __bf16* Wr_t = (__bf16*)alloc((size_t)C_ * C_ * 2);
  __bf16* Wk_t = (__bf16*)alloc((size_t)C_ * C_ * 2);
  __bf16* Wv_t = (__bf16*)alloc((size_t)C_ * C_ * 2);
  __bf16* Wo_t = (__bf16*)alloc((size_t)C_ * C_ * 2);
  __bf16* ww1_t = (__bf16*)alloc((size_t)C_ * D_DECAY * 2);
  __bf16* ww2_t = (__bf16*)alloc((size_t)C_ * D_DECAY * 2);
  __bf16* aa1_t = (__bf16*)alloc((size_t)C_ * D_AAA * 2);
  __bf16* aa2_t = (__bf16*)alloc((size_t)C_ * D_AAA * 2);
  __bf16* vv1_t = (__bf16*)alloc((size_t)C_ * D_MV * 2);
  __bf16* vv2_t = (__bf16*)alloc((size_t)C_ * D_MV * 2);
  __bf16* gg1_t = (__bf16*)alloc((size_t)C_ * D_GATE * 2);
  __bf16* gg2_t = (__bf16*)alloc((size_t)C_ * D_GATE * 2);

  __bf16* xr = (__bf16*)alloc((size_t)BTC_ * 2);
  __bf16* xw = (__bf16*)alloc((size_t)BTC_ * 2);
  __bf16* xk = (__bf16*)alloc((size_t)BTC_ * 2);
  __bf16* xv = (__bf16*)alloc((size_t)BTC_ * 2);
  __bf16* xa = (__bf16*)alloc((size_t)BTC_ * 2);
  __bf16* xg = (__bf16*)alloc((size_t)BTC_ * 2);

  float* rbuf  = (float*)alloc((size_t)BTC_ * 4);
  float* kbuf  = (float*)alloc((size_t)BTC_ * 4);
  float* vbuf  = (float*)alloc((size_t)BTC_ * 4);
  float* dw    = (float*)alloc((size_t)BTC_ * 4);
  float* da    = (float*)alloc((size_t)BTC_ * 4);
  float* dv    = (float*)alloc((size_t)BTC_ * 4);
  float* gatef = (float*)alloc((size_t)BTC_ * 4);
  float* decay = (float*)alloc((size_t)BTC_ * 4);
  float* arec  = (float*)alloc((size_t)BTC_ * 4);
  float* brec  = (float*)alloc((size_t)BTC_ * 4);
  float* ybuf  = (float*)alloc((size_t)BTC_ * 4);

  float* hw = (float*)alloc((size_t)BT_ * D_DECAY * 4);
  float* ha = (float*)alloc((size_t)BT_ * D_AAA * 4);
  float* hv = (float*)alloc((size_t)BT_ * D_MV * 4);
  float* hg = (float*)alloc((size_t)BT_ * D_GATE * 4);
  __bf16* hwb = (__bf16*)alloc((size_t)BT_ * D_DECAY * 2);
  __bf16* hab = (__bf16*)alloc((size_t)BT_ * D_AAA * 2);
  __bf16* hvb = (__bf16*)alloc((size_t)BT_ * D_MV * 2);
  __bf16* hgb = (__bf16*)alloc((size_t)BT_ * D_GATE * 2);
  __bf16* ygb = (__bf16*)alloc((size_t)BTC_ * 2);

  // 1) weights -> bf16 transposed [N,K]
  auto tr = [&](const float* in, __bf16* out, int K, int N) {
    int tot = K * N;
    k_transpose_bf16<<<dim3((tot + 255) / 256), dim3(256), 0, stream>>>(in, out, K, N);
  };
  tr(W_r, Wr_t, C_, C_);
  tr(W_k, Wk_t, C_, C_);
  tr(W_v, Wv_t, C_, C_);
  tr(W_o, Wo_t, C_, C_);
  tr(ww_w1, ww1_t, C_, D_DECAY);
  tr(ww_w2, ww2_t, D_DECAY, C_);
  tr(aa_w1, aa1_t, C_, D_AAA);
  tr(aa_w2, aa2_t, D_AAA, C_);
  tr(vv_w1, vv1_t, C_, D_MV);
  tr(vv_w2, vv2_t, D_MV, C_);
  tr(gg_w1, gg1_t, C_, D_GATE);
  tr(gg_w2, gg2_t, D_GATE, C_);

  // 2) token-shift mix -> bf16 activations
  k_mix<<<dim3((BTC_ + 255) / 256), dim3(256), 0, stream>>>(
      x, xx_r, xx_w, xx_k, xx_v, xx_a, xx_g, xr, xw, xk, xv, xa, xg);

  // wide (32x64/wave) GEMM for N % 64 == 0; narrow (32x32/wave) otherwise
  auto gemm_w = [&](const __bf16* A, const __bf16* Bt, float* Cm, int M, int N, int K) {
    k_gemm_bf16_2x4<<<dim3(N / 64, M / 32), dim3(32), 0, stream>>>(A, Bt, Cm, M, N, K);
  };
  auto gemm_n = [&](const __bf16* A, const __bf16* Bt, float* Cm, int M, int N, int K) {
    k_gemm_bf16_32<<<dim3(N / 32, M / 32), dim3(32), 0, stream>>>(A, Bt, Cm, M, N, K);
  };

  // 3) big projections + low-rank stage 1
  gemm_w(xr, Wr_t, rbuf, BT_, C_, C_);
  gemm_w(xk, Wk_t, kbuf, BT_, C_, C_);
  gemm_w(xv, Wv_t, vbuf, BT_, C_, C_);
  gemm_n(xw, ww1_t, hw, BT_, D_DECAY, C_);
  gemm_n(xa, aa1_t, ha, BT_, D_AAA, C_);
  gemm_n(xv, vv1_t, hv, BT_, D_MV, C_);
  gemm_n(xg, gg1_t, hg, BT_, D_GATE, C_);

  // 4) hidden activations -> bf16
  k_act_bf16<1><<<dim3((BT_ * D_DECAY + 255) / 256), 256, 0, stream>>>(hw, hwb, BT_ * D_DECAY);
  k_act_bf16<0><<<dim3((BT_ * D_AAA + 255) / 256), 256, 0, stream>>>(ha, hab, BT_ * D_AAA);
  k_act_bf16<0><<<dim3((BT_ * D_MV + 255) / 256), 256, 0, stream>>>(hv, hvb, BT_ * D_MV);
  k_act_bf16<2><<<dim3((BT_ * D_GATE + 255) / 256), 256, 0, stream>>>(hg, hgb, BT_ * D_GATE);

  // 5) low-rank stage 2 (all N = C_, wide kernel)
  gemm_w(hwb, ww2_t, dw, BT_, C_, D_DECAY);
  gemm_w(hab, aa2_t, da, BT_, C_, D_AAA);
  gemm_w(hvb, vv2_t, dv, BT_, C_, D_MV);
  gemm_w(hgb, gg2_t, gatef, BT_, C_, D_GATE);

  // 6) per-(token,head) post-processing
  k_post<<<dim3(BT_ * H_), dim3(64), 0, stream>>>(
      v0, dw, da, dv, ww_b, aa_b, vv_b, kk_s, ka_s, kbuf, vbuf, decay, arec, brec);

  // 7) sequential RWKV7 state recurrence
  k_rwkv7<<<dim3(B_ * H_), dim3(64), 0, stream>>>(rbuf, decay, kbuf, vbuf, arec, brec, ybuf);

  // 8) groupnorm + bonus + gate -> bf16
  k_gn_bonus_gate<<<dim3(BT_ * H_), dim3(64), 0, stream>>>(
      ybuf, rbuf, kbuf, vbuf, gatef, rk_s, gn_w, gn_b, ygb);

  // 9) output projection -> d_out[0:BTC], and v0 passthrough -> d_out[BTC:2*BTC]
  float* out0 = (float*)d_out;
  gemm_w(ygb, Wo_t, out0, BT_, C_, C_);
  k_copy_f32<<<dim3((BTC_ + 255) / 256), 256, 0, stream>>>(v0, out0 + BTC_, BTC_);
}
